// BidirectionalMamba_36644660969831
// MI455X (gfx1250) — compile-verified, matrix operand order checked by reading, not run on hardware
//
#include <hip/hip_runtime.h>
#include <hip/hip_bf16.h>
#include <math.h>

// ---------------------------------------------------------------------------
// BiMamba on MI455X (gfx1250): all matmuls via v_wmma_f32_16x16x32_bf16 with
// TDM (tensor_load_to_lds) staging of the A slab; selective scan with the
// 16-state recurrence held in VGPRs (lane per channel).
// ---------------------------------------------------------------------------

#define B_  4
#define L_  2048
#define T_  (B_ * L_)
#define DM_ 256          // d_model
#define DI_ 512          // d_inner
#define DS_ 16           // d_state
#define DTR_ 16          // dt_rank
#define NXP_ 64          // x_proj out cols padded 48 -> 64

typedef __bf16 bf16;
typedef __attribute__((ext_vector_type(16))) __bf16 v16bf;
typedef __attribute__((ext_vector_type(4)))  __bf16 v4bf;
typedef __attribute__((ext_vector_type(8)))  float  v8f;
typedef __attribute__((ext_vector_type(4)))  unsigned int v4u;
typedef __attribute__((ext_vector_type(8)))  int    v8i;
typedef __attribute__((ext_vector_type(4)))  int    v4i;

#if __has_builtin(__builtin_amdgcn_tensor_load_to_lds) && \
    __has_builtin(__builtin_amdgcn_s_wait_tensorcnt)
#define USE_TDM 1
#else
#define USE_TDM 0
#endif

#if USE_TDM
// LDS byte offset of a __shared__ object (generic -> AS3 -> int strips aperture)
__device__ __forceinline__ unsigned lds_byte_off(void* p) {
  return (unsigned)(unsigned long long)(__attribute__((address_space(3))) void*)p;
}

// 2-D TDM load: tile (rows x cols) of a row-major (tdim1 x tdim0) bf16 tensor
// into LDS at lds_addr, written contiguously row-major.
__device__ __forceinline__ void tdm_load_tile_bf16(
    unsigned lds_addr, const void* gaddr,
    unsigned cols /*tile_dim0, elems*/, unsigned rows /*tile_dim1*/,
    unsigned tdim0 /*K*/, unsigned tdim1 /*M*/, unsigned stride0 /*elems*/)
{
  unsigned long long ga = (unsigned long long)gaddr;
  v4u g0;
  g0[0] = 1u;                                               // count=1
  g0[1] = lds_addr;                                         // lds_addr
  g0[2] = (unsigned)(ga & 0xFFFFFFFFu);                     // global addr lo
  g0[3] = (unsigned)((ga >> 32) & 0x01FFFFFFu) | (2u << 30);// addr hi | type=2
  v8i g1;
  g1[0] = (int)(1u << 16);                                  // data_size=1 (2B)
  g1[1] = (int)((tdim0 & 0xFFFFu) << 16);                   // tensor_dim0 lo16
  g1[2] = (int)(((tdim0 >> 16) & 0xFFFFu) |                 // tensor_dim0 hi16
                ((tdim1 & 0xFFFFu) << 16));                 // tensor_dim1 lo16
  g1[3] = (int)(((tdim1 >> 16) & 0xFFFFu) |                 // tensor_dim1 hi16
                ((cols & 0xFFFFu) << 16));                  // tile_dim0
  g1[4] = (int)(rows & 0xFFFFu);                            // tile_dim1; tile_dim2=0
  g1[5] = (int)stride0;                                     // tensor_dim0_stride lo32
  g1[6] = 0;                                                // stride hi / dim1_stride
  g1[7] = 0;
  v4i g2 = {0, 0, 0, 0};
  v4i g3 = {0, 0, 0, 0};
#if __clang_major__ >= 23
  v8i g4 = {0, 0, 0, 0, 0, 0, 0, 0};
  __builtin_amdgcn_tensor_load_to_lds(g0, g1, g2, g3, g4, 0);
#else
  __builtin_amdgcn_tensor_load_to_lds(g0, g1, g2, g3, 0);
#endif
}
#endif // USE_TDM

// ---------------------------------------------------------------------------
// WMMA GEMM:  C[M,N] = act( A[M,K](bf16) * W[N,K]^T(bf16) + bias + addend )
// Block = NW waves. Each wave owns a 16(M) x 64(N) tile: 4 accumulators,
// 4 WMMAs per K-step, A fragment reused. The 16xK A-slab for the block is
// staged into LDS once by a single TDM descriptor (wave 0), then the K-loop
// runs barrier-free. ACT: 0=none 1=silu 2=softplus.
// ---------------------------------------------------------------------------
template <int ACT, int NW>
__global__ __launch_bounds__(NW * 32) void k_gemm_wmma(
    const bf16* __restrict__ A, const bf16* __restrict__ W,
    const float* __restrict__ bias, const float* __restrict__ addend,
    float* __restrict__ C, int M, int N, int K)
{
  __shared__ bf16 Alds[16 * 768];          // max K = 768 -> 24 KB

  const int wave = threadIdx.x >> 5;
  const int lane = threadIdx.x & 31;
  const int m0 = blockIdx.y << 4;
  const int n0 = (blockIdx.x * NW + wave) << 6;   // 64-col tile per wave

#if USE_TDM
  if (wave == 0) {
    tdm_load_tile_bf16(lds_byte_off(&Alds[0]), &A[(size_t)m0 * K],
                       (unsigned)K, 16u, (unsigned)K, (unsigned)M, (unsigned)K);
    __builtin_amdgcn_s_wait_tensorcnt(0);
  }
#else
  // cooperative copy (4 bf16 per chunk)
  for (int i = threadIdx.x * 4; i < 16 * K; i += NW * 32 * 4) {
    const int row = i / K;
    const int ck  = i - row * K;
    *(v4bf*)&Alds[i] = *(const v4bf*)&A[(size_t)(m0 + row) * K + ck];
  }
#endif
  __syncthreads();

  const int r    = lane & 15;
  const int koff = (lane >> 4) << 4;       // 0 or 16

  v8f acc[4] = {v8f{}, v8f{}, v8f{}, v8f{}};
  for (int k0 = 0; k0 < K; k0 += 32) {
    v16bf af = *(const v16bf*)&Alds[r * K + k0 + koff];
    const bf16* wp = &W[(size_t)(n0 + r) * K + (k0 + koff)];
    v16bf bfr[4];
#pragma unroll
    for (int j = 0; j < 4; ++j) bfr[j] = *(const v16bf*)(wp + (size_t)(16 * j) * K);
#pragma unroll
    for (int j = 0; j < 4; ++j)
      acc[j] = __builtin_amdgcn_wmma_f32_16x16x32_bf16(
          false, af, false, bfr[j], (short)0, acc[j], false, false);
  }

  // Epilogue per ISA C/D layout: VGPR v -> row v (lanes 0-15) / v+8 (16-31)
  const int rbase = m0 + ((lane >> 4) << 3);
#pragma unroll
  for (int j = 0; j < 4; ++j) {
    const int col = n0 + j * 16 + (lane & 15);
    const float bia = bias ? bias[col] : 0.0f;
#pragma unroll
    for (int v = 0; v < 8; ++v) {
      const int row = rbase + v;
      float val = acc[j][v] + bia;
      if (addend) val += addend[(size_t)row * N + col];
      if (ACT == 1) val = val / (1.0f + __expf(-val));                 // silu
      if (ACT == 2) val = (val > 20.0f) ? val : log1pf(__expf(val));   // softplus
      C[(size_t)row * N + col] = val;
    }
  }
}

// ---------------------------------------------------------------------------
// RMSNorm (256 cols), one wave per row, bf16 output for the next GEMM.
// ---------------------------------------------------------------------------
__global__ __launch_bounds__(32) void k_rmsnorm_bf16(
    const float* __restrict__ x, const float* __restrict__ w,
    bf16* __restrict__ out)
{
  const size_t row = blockIdx.x;
  const float4* xr = (const float4*)(x + row * DM_);
  const int lane = threadIdx.x;
  float4 a = xr[lane];
  float4 c = xr[lane + 32];
  float ss = a.x*a.x + a.y*a.y + a.z*a.z + a.w*a.w
           + c.x*c.x + c.y*c.y + c.z*c.z + c.w*c.w;
#pragma unroll
  for (int off = 16; off > 0; off >>= 1) ss += __shfl_xor(ss, off, 32);
  const float rs = rsqrtf(ss * (1.0f / DM_) + 1e-5f);
  bf16* o = out + row * DM_;
  const int c0 = lane * 4;
  o[c0+0]     = (bf16)(a.x * rs * w[c0+0]);
  o[c0+1]     = (bf16)(a.y * rs * w[c0+1]);
  o[c0+2]     = (bf16)(a.z * rs * w[c0+2]);
  o[c0+3]     = (bf16)(a.w * rs * w[c0+3]);
  o[c0+128+0] = (bf16)(c.x * rs * w[c0+128+0]);
  o[c0+128+1] = (bf16)(c.y * rs * w[c0+128+1]);
  o[c0+128+2] = (bf16)(c.z * rs * w[c0+128+2]);
  o[c0+128+3] = (bf16)(c.w * rs * w[c0+128+3]);
}

// Depthwise causal conv (K=3) over xi = xz[:, :512], then SiLU, bf16 out.
__global__ void k_dwconv_silu(const float* __restrict__ xz,
                              const float* __restrict__ cw,
                              const float* __restrict__ cb,
                              bf16* __restrict__ out)
{
  int idx = blockIdx.x * blockDim.x + threadIdx.x;     // over T_*DI_
  if (idx >= T_ * DI_) return;
  const int c = idx & (DI_ - 1);
  const int t = idx >> 9;
  const int l = t & (L_ - 1);
  const float w0 = cw[c*3+0], w1 = cw[c*3+1], w2 = cw[c*3+2];
  float x2 = xz[(size_t)t * 1024 + c];
  float x1 = (l >= 1) ? xz[(size_t)(t-1) * 1024 + c] : 0.0f;
  float x0 = (l >= 2) ? xz[(size_t)(t-2) * 1024 + c] : 0.0f;
  float v = w0*x0 + w1*x1 + w2*x2 + cb[c];
  v = v / (1.0f + __expf(-v));
  out[idx] = (bf16)v;
}

// dt (first 16 cols of x_dbl) -> bf16, zero-padded to K=32 for WMMA depth.
__global__ void k_dtpad(const float* __restrict__ xdbl, bf16* __restrict__ out)
{
  int idx = blockIdx.x * blockDim.x + threadIdx.x;     // over T_*32
  if (idx >= T_ * 32) return;
  const int c = idx & 31;
  const int t = idx >> 5;
  out[idx] = (c < DTR_) ? (bf16)xdbl[(size_t)t * NXP_ + c] : (bf16)0.0f;
}

// ---------------------------------------------------------------------------
// Selective scan: lane per (b,d). h[16] and A_row[16] live in VGPRs; the
// shared B/C vectors are staged through LDS in 16-step chunks.
// ---------------------------------------------------------------------------
__global__ __launch_bounds__(DI_) void k_scan(
    const float* __restrict__ delta, const bf16* __restrict__ u_bf,
    const float* __restrict__ xdbl, const float* __restrict__ A_log,
    const float* __restrict__ Dp, float* __restrict__ y)
{
  const int b = blockIdx.x;
  const int d = threadIdx.x;
  float Ar[DS_];
#pragma unroll
  for (int n = 0; n < DS_; ++n) Ar[n] = -__expf(A_log[d * DS_ + n]);
  const float Dd = Dp[d];
  float h[DS_];
#pragma unroll
  for (int n = 0; n < DS_; ++n) h[n] = 0.0f;

  __shared__ float BC[16][32];      // [step][Bm(16) | Cm(16)]
  const size_t base = (size_t)b * L_;
  for (int l0 = 0; l0 < L_; l0 += 16) {
    __syncthreads();
    {
      const int tt = threadIdx.x >> 5;
      const int cc = threadIdx.x & 31;
      BC[tt][cc] = xdbl[(base + l0 + tt) * NXP_ + DTR_ + cc];
    }
    __syncthreads();
    for (int tt = 0; tt < 16; ++tt) {
      const size_t t = base + l0 + tt;
      const float dl = delta[t * DI_ + d];
      const float uu = (float)u_bf[t * DI_ + d];
      const float du = dl * uu;
      float acc = 0.0f;
#pragma unroll
      for (int n = 0; n < DS_; ++n) {
        const float dA = __expf(dl * Ar[n]);
        h[n] = dA * h[n] + du * BC[tt][n];
        acc += h[n] * BC[tt][DS_ + n];
      }
      y[t * DI_ + d] = acc + uu * Dd;
    }
  }
}

// y * silu(z), z = xz[:, 512:1024]; bf16 out for out_proj GEMM.
__global__ void k_zmul(const float* __restrict__ yscan,
                       const float* __restrict__ xz, bf16* __restrict__ out)
{
  int idx = blockIdx.x * blockDim.x + threadIdx.x;     // over T_*DI_
  if (idx >= T_ * DI_) return;
  const int c = idx & (DI_ - 1);
  const int t = idx >> 9;
  const float z = xz[(size_t)t * 1024 + DI_ + c];
  out[idx] = (bf16)(yscan[idx] * (z / (1.0f + __expf(-z))));
}

// im2col for the branch conv (pad=1): row t = [x[t-1] | x[t] | x[t+1]] (bf16).
__global__ void k_im2col(const float* __restrict__ hid, bf16* __restrict__ out)
{
  int idx = blockIdx.x * blockDim.x + threadIdx.x;     // over T_*768
  if (idx >= T_ * 768) return;
  const int t   = idx / 768;
  const int c   = idx - t * 768;
  const int seg = c >> 8;          // 0,1,2 -> taps at t-1, t, t+1
  const int ch  = c & 255;
  const int l   = t & (L_ - 1);
  const int lo  = l + seg - 1;
  bf16 v = (bf16)0.0f;
  if (lo >= 0 && lo < L_) v = (bf16)hid[(size_t)(t + seg - 1) * DM_ + ch];
  out[idx] = v;
}

__global__ void k_flip(const float* __restrict__ in, float* __restrict__ out)
{
  int idx = blockIdx.x * blockDim.x + threadIdx.x;     // over T_*DM_
  if (idx >= T_ * DM_) return;
  const int c = idx & (DM_ - 1);
  const int t = idx >> 8;
  const int b = t / L_, l = t & (L_ - 1);
  out[idx] = in[((size_t)(b * L_ + (L_ - 1 - l)) << 8) + c];
}

// x_next = res_fwd + flip(res_bwd) + x_in
__global__ void k_combine(const float* __restrict__ rf,
                          const float* __restrict__ rb,
                          const float* __restrict__ xin,
                          float* __restrict__ out)
{
  int idx = blockIdx.x * blockDim.x + threadIdx.x;     // over T_*DM_
  if (idx >= T_ * DM_) return;
  const int c = idx & (DM_ - 1);
  const int t = idx >> 8;
  const int b = t / L_, l = t & (L_ - 1);
  const size_t j = (((size_t)(b * L_ + (L_ - 1 - l))) << 8) + c;
  out[idx] = rf[idx] + rb[j] + xin[idx];
}

// weight casts -----------------------------------------------------------
__global__ void k_cast(const float* __restrict__ in, bf16* __restrict__ out, int n)
{
  int i = blockIdx.x * blockDim.x + threadIdx.x;
  if (i < n) out[i] = (bf16)in[i];
}
// x_proj (48x512) -> (64x512) zero-padded rows
__global__ void k_cast_xw(const float* __restrict__ w, bf16* __restrict__ out)
{
  int i = blockIdx.x * blockDim.x + threadIdx.x;       // over 64*512
  if (i >= NXP_ * DI_) return;
  const int n = i >> 9, k = i & (DI_ - 1);
  out[i] = (n < 48) ? (bf16)w[n * DI_ + k] : (bf16)0.0f;
}
// dt_proj_w (512x16) -> (512x32) zero padded
__global__ void k_cast_dtw(const float* __restrict__ w, bf16* __restrict__ out)
{
  int i = blockIdx.x * blockDim.x + threadIdx.x;       // over 512*32
  if (i >= DI_ * 32) return;
  const int n = i >> 5, k = i & 31;
  out[i] = (k < DTR_) ? (bf16)w[n * DTR_ + k] : (bf16)0.0f;
}
// conv_w (O=256, I=256, K=3) -> row o = [w[o,:,0] | w[o,:,1] | w[o,:,2]]
__global__ void k_cast_convw(const float* __restrict__ w, bf16* __restrict__ out)
{
  int i = blockIdx.x * blockDim.x + threadIdx.x;       // over 256*768
  if (i >= DM_ * 768) return;
  const int o  = i / 768;
  const int c  = i - o * 768;
  const int k  = c >> 8;
  const int ch = c & 255;
  out[i] = (bf16)w[((size_t)o * DM_ + ch) * 3 + k];
}

// ---------------------------------------------------------------------------
// Host orchestration
// ---------------------------------------------------------------------------
struct BlockW {
  const float *A_log, *D, *conv_b, *conv_w, *dt_proj_b, *dt_proj_w,
              *in_proj, *out_proj, *x_proj, *mlp_b, *mlp_w, *norm2_w, *norm_w;
};
struct Ws {
  float *resid, *hidden, *xz, *delta, *yscan, *xdbl;
  bf16  *hn_bf, *xi_bf, *dt_bf, *ym_bf, *im2col;
  bf16  *w_in, *w_x, *w_dt, *w_out, *w_mlp, *w_conv;
};

static inline int cdiv(int a, int b) { return (a + b - 1) / b; }

static void run_branch(const float* xin, const BlockW& bw,
                       const float* conv_w_br, const float* conv_b_br,
                       float* res_out, const Ws& w, hipStream_t s)
{
  // cast weights to bf16 (stream-ordered; deterministic)
  k_cast      <<<cdiv(2*DI_*DM_, 256), 256, 0, s>>>(bw.in_proj,  w.w_in,  2*DI_*DM_);
  k_cast_xw   <<<cdiv(NXP_*DI_,  256), 256, 0, s>>>(bw.x_proj,   w.w_x);
  k_cast_dtw  <<<cdiv(DI_*32,    256), 256, 0, s>>>(bw.dt_proj_w, w.w_dt);
  k_cast      <<<cdiv(DM_*DI_,   256), 256, 0, s>>>(bw.out_proj, w.w_out, DM_*DI_);
  k_cast      <<<cdiv(DM_*DM_,   256), 256, 0, s>>>(bw.mlp_w,    w.w_mlp, DM_*DM_);
  k_cast_convw<<<cdiv(DM_*768,   256), 256, 0, s>>>(conv_w_br,   w.w_conv);

  // residual = hidden (first block), then the Mamba block
  hipMemcpyAsync(w.resid, xin, (size_t)T_ * DM_ * 4, hipMemcpyDeviceToDevice, s);
  k_rmsnorm_bf16<<<T_, 32, 0, s>>>(w.resid, bw.norm_w, w.hn_bf);

  // xz = hn @ in_proj^T   (M=8192, N=1024, K=256); wave tile 16x64, 4 waves
  k_gemm_wmma<0,4><<<dim3(4, T_/16), 128, 0, s>>>(w.hn_bf, w.w_in, nullptr, nullptr,
                                                  w.xz, T_, 1024, DM_);
  k_dwconv_silu<<<cdiv(T_*DI_, 256), 256, 0, s>>>(w.xz, bw.conv_w, bw.conv_b, w.xi_bf);

  // x_dbl = xi @ x_proj^T (N padded to 64 -> single 16x64 wave tile)
  k_gemm_wmma<0,1><<<dim3(1, T_/16), 32, 0, s>>>(w.xi_bf, w.w_x, nullptr, nullptr,
                                                 w.xdbl, T_, NXP_, DI_);
  k_dtpad<<<cdiv(T_*32, 256), 256, 0, s>>>(w.xdbl, w.dt_bf);

  // delta = softplus(dt @ dt_proj^T + b)  (K padded to 32)
  k_gemm_wmma<2,4><<<dim3(2, T_/16), 128, 0, s>>>(w.dt_bf, w.w_dt, bw.dt_proj_b, nullptr,
                                                  w.delta, T_, DI_, 32);
  k_scan<<<B_, DI_, 0, s>>>(w.delta, w.xi_bf, w.xdbl, bw.A_log, bw.D, w.yscan);
  k_zmul<<<cdiv(T_*DI_, 256), 256, 0, s>>>(w.yscan, w.xz, w.ym_bf);

  // residual = y @ out_proj^T + residual
  k_gemm_wmma<0,4><<<dim3(1, T_/16), 128, 0, s>>>(w.ym_bf, w.w_out, nullptr, w.resid,
                                                  w.resid, T_, DM_, DI_);
  k_rmsnorm_bf16<<<T_, 32, 0, s>>>(w.resid, bw.norm2_w, w.hn_bf);
  // hidden = hn2 @ mlp^T + mlp_b
  k_gemm_wmma<0,4><<<dim3(1, T_/16), 128, 0, s>>>(w.hn_bf, w.w_mlp, bw.mlp_b, nullptr,
                                                  w.hidden, T_, DM_, DM_);
  // branch conv1d (pad=1) as im2col GEMM, fused "+ residual"
  k_im2col<<<cdiv(T_*768, 256), 256, 0, s>>>(w.hidden, w.im2col);
  k_gemm_wmma<0,4><<<dim3(1, T_/16), 128, 0, s>>>(w.im2col, w.w_conv, conv_b_br, w.resid,
                                                  res_out, T_, DM_, 768);
}

extern "C" void kernel_launch(void* const* d_in, const int* in_sizes, int n_in,
                              void* d_out, int out_size, void* d_ws, size_t ws_size,
                              hipStream_t stream)
{
  (void)in_sizes; (void)n_in; (void)out_size; (void)ws_size;

  // workspace carve-out (~157 MB)
  char* p = (char*)d_ws;
  auto alloc = [&](size_t bytes) -> void* {
    void* r = (void*)p; p += (bytes + 255) & ~(size_t)255; return r;
  };
  Ws w;
  float* x_mid   = (float*)alloc((size_t)T_ * DM_ * 4);
  float* xflip   = (float*)alloc((size_t)T_ * DM_ * 4);
  float* res_fwd = (float*)alloc((size_t)T_ * DM_ * 4);
  float* res_bwd = (float*)alloc((size_t)T_ * DM_ * 4);
  w.resid  = (float*)alloc((size_t)T_ * DM_  * 4);
  w.hidden = (float*)alloc((size_t)T_ * DM_  * 4);
  w.xz     = (float*)alloc((size_t)T_ * 1024 * 4);
  w.delta  = (float*)alloc((size_t)T_ * DI_  * 4);
  w.yscan  = (float*)alloc((size_t)T_ * DI_  * 4);
  w.xdbl   = (float*)alloc((size_t)T_ * NXP_ * 4);
  w.hn_bf  = (bf16*)alloc((size_t)T_ * DM_ * 2);
  w.xi_bf  = (bf16*)alloc((size_t)T_ * DI_ * 2);
  w.dt_bf  = (bf16*)alloc((size_t)T_ * 32  * 2);
  w.ym_bf  = (bf16*)alloc((size_t)T_ * DI_ * 2);
  w.im2col = (bf16*)alloc((size_t)T_ * 768 * 2);
  w.w_in   = (bf16*)alloc((size_t)2*DI_*DM_ * 2);
  w.w_x    = (bf16*)alloc((size_t)NXP_*DI_  * 2);
  w.w_dt   = (bf16*)alloc((size_t)DI_*32    * 2);
  w.w_out  = (bf16*)alloc((size_t)DM_*DI_   * 2);
  w.w_mlp  = (bf16*)alloc((size_t)DM_*DM_   * 2);
  w.w_conv = (bf16*)alloc((size_t)DM_*768   * 2);

  // input layout: [x, layer0..layer1], each layer (pytree sorted keys):
  // bwd_block(13), bwd_conv_b, bwd_conv_w, fwd_block(13), fwd_conv_b, fwd_conv_w
  // block (sorted): mixer(9: A_log,D,conv_b,conv_w,dt_proj_b,dt_proj_w,in_proj,
  //                 out_proj,x_proj), mlp_b, mlp_w, norm2_w, norm_w
  auto get = [&](int i) { return (const float*)d_in[i]; };
  auto load_block = [&](int base) {
    BlockW b;
    b.A_log = get(base+0);  b.D = get(base+1);  b.conv_b = get(base+2);
    b.conv_w = get(base+3); b.dt_proj_b = get(base+4); b.dt_proj_w = get(base+5);
    b.in_proj = get(base+6); b.out_proj = get(base+7); b.x_proj = get(base+8);
    b.mlp_b = get(base+9); b.mlp_w = get(base+10);
    b.norm2_w = get(base+11); b.norm_w = get(base+12);
    return b;
  };

  const float* xcur = (const float*)d_in[0];
  const int N_LAYERS = 2;
  for (int layer = 0; layer < N_LAYERS; ++layer) {
    const int base = 1 + layer * 30;
    BlockW bwd = load_block(base + 0);
    const float* bwd_conv_b = get(base + 13);
    const float* bwd_conv_w = get(base + 14);
    BlockW fwd = load_block(base + 15);
    const float* fwd_conv_b = get(base + 28);
    const float* fwd_conv_w = get(base + 29);

    // forward branch
    run_branch(xcur, fwd, fwd_conv_w, fwd_conv_b, res_fwd, w, stream);
    // backward branch on flipped sequence
    k_flip<<<cdiv(T_*DM_, 256), 256, 0, stream>>>(xcur, xflip);
    run_branch(xflip, bwd, bwd_conv_w, bwd_conv_b, res_bwd, w, stream);

    float* xnext = (layer == N_LAYERS - 1) ? (float*)d_out : x_mid;
    k_combine<<<cdiv(T_*DM_, 256), 256, 0, stream>>>(res_fwd, res_bwd, xcur, xnext);
    xcur = xnext;
  }
}
